// GraphDecoder_14620068675789
// MI455X (gfx1250) — matrix operand order repaired and checked by reading, not compiled, not run on hardware
//
#include <hip/hip_runtime.h>

// Problem constants (from the reference)
#define B_   256
#define LAT_ 64
#define N_   128
#define E_   512
#define BN_  (B_ * N_)   // 32768
#define BE_  (B_ * E_)   // 131072

typedef __attribute__((ext_vector_type(16))) _Float16 v16h;
typedef __attribute__((ext_vector_type(8)))  _Float16 h8;
typedef __attribute__((ext_vector_type(4)))  _Float16 h4;
typedef __attribute__((ext_vector_type(8)))  float    v8f;
typedef __attribute__((ext_vector_type(4)))  float    f4;
typedef __attribute__((ext_vector_type(4)))  unsigned int u32x4;
typedef __attribute__((ext_vector_type(4)))  int      i32x4;
typedef __attribute__((ext_vector_type(8)))  int      i32x8;

#if defined(__HIP_DEVICE_COMPILE__) && __has_builtin(__builtin_amdgcn_tensor_load_to_lds) && \
    __has_builtin(__builtin_amdgcn_s_wait_tensorcnt)
#define USE_TDM 1
#else
#define USE_TDM 0
#endif

#if USE_TDM
// Issue a TDM DMA of one 32x128 f32 tile of W3 (row stride 65536 elements)
// into LDS at byte offset lds_bytes, with LDS padding: +4 dwords every 128
// dwords (so the LDS image has row stride 132 dwords, matching sBf).
__device__ __forceinline__ void tdm_load_tile(const float* gsrc, unsigned lds_bytes) {
    unsigned long long ga = (unsigned long long)(const void*)gsrc;
    u32x4 g0 = { 1u,                                   // count=1 (valid D#)
                 lds_bytes,                            // lds_addr (bytes)
                 (unsigned)(ga & 0xffffffffu),         // global_addr[31:0]
                 (unsigned)((ga >> 32) & 0x01ffffffu)  // global_addr[56:32]
                     | 0x80000000u };                  // type=2 ("image")
    // group1: data_size=4B, pad_enable, pad_interval=128dw, pad_amount=4dw
    i32x8 g1 = { (int)0x07920000,
                 0,                                    // tensor_dim0 lo16 (=0)
                 (int)((65536u >> 16) | (256u << 16)), // dim0 hi | dim1 lo
                 (int)(128u << 16),                    // dim1 hi | tile_dim0=128
                 32,                                   // tile_dim1=32, tile_dim2=0
                 65536,                                // dim0_stride lo32
                 0, 0 };                               // stride hi / dim1_stride
    i32x4 z4 = { 0, 0, 0, 0 };
#if defined(__clang_major__) && (__clang_major__ >= 23)
    i32x8 z8 = { 0, 0, 0, 0, 0, 0, 0, 0 };
    __builtin_amdgcn_tensor_load_to_lds(g0, g1, z4, z4, z8, 0);
#else
    __builtin_amdgcn_tensor_load_to_lds(g0, g1, z4, z4, 0);
#endif
}
#endif

// ---------------------------------------------------------------------------
// Generic WMMA GEMM:  C(M,N) = act(A(M,K) @ W(K,N) + bias)
// Requirements: M % 32 == 0, K % 32 == 0 (callers pad K; pad columns of A and
// pad rows of W are zero).  N arbitrary (vector fast path when N % 32 == 0).
// Block = 128 threads = 4 wave32's; each wave owns one 16x16 tile of a 32x32
// block tile.  Staging is manually unrolled so all b128 loads are in flight
// before any wait; f32 -> f16 conversion happens during staging; f32 accum.
// ---------------------------------------------------------------------------
__global__ void k_gemm(const float* __restrict__ A, const float* __restrict__ W,
                       const float* __restrict__ bias, float* __restrict__ C,
                       int M, int N, int K, int relu)
{
    (void)M;
    __shared__ _Float16 sA[32][40];    // [m][k], row stride 80B (16B aligned chunks)
    __shared__ _Float16 sBt[32][40];   // [n][k] (transposed W tile)
    const int bm   = blockIdx.y * 32;
    const int bn   = blockIdx.x * 32;
    const int tid  = threadIdx.x;
    const int wave = tid >> 5;
    const int lane = tid & 31;
    const bool nfull = ((N & 31) == 0);
    const int r0 = tid >> 3, c4 = (tid & 7) << 2;   // staging slot coords

    v8f acc = {};
    for (int k0 = 0; k0 < K; k0 += 32) {
        // ---- stage A (32x32) + W (32x32): overlap the 4 b128 loads ----
        f4 va0 = *(const f4*)&A[(size_t)(bm + r0) * K + k0 + c4];
        f4 va1 = *(const f4*)&A[(size_t)(bm + r0 + 16) * K + k0 + c4];
        if (nfull) {
            f4 vb0 = *(const f4*)&W[(size_t)(k0 + r0) * N + bn + c4];
            f4 vb1 = *(const f4*)&W[(size_t)(k0 + r0 + 16) * N + bn + c4];
            h4 h0 = { (_Float16)va0[0], (_Float16)va0[1], (_Float16)va0[2], (_Float16)va0[3] };
            h4 h1 = { (_Float16)va1[0], (_Float16)va1[1], (_Float16)va1[2], (_Float16)va1[3] };
            *(h4*)&sA[r0][c4]      = h0;
            *(h4*)&sA[r0 + 16][c4] = h1;
            #pragma unroll
            for (int q = 0; q < 4; ++q) {
                sBt[c4 + q][r0]      = (_Float16)vb0[q];
                sBt[c4 + q][r0 + 16] = (_Float16)vb1[q];
            }
        } else {
            h4 h0 = { (_Float16)va0[0], (_Float16)va0[1], (_Float16)va0[2], (_Float16)va0[3] };
            h4 h1 = { (_Float16)va1[0], (_Float16)va1[1], (_Float16)va1[2], (_Float16)va1[3] };
            *(h4*)&sA[r0][c4]      = h0;
            *(h4*)&sA[r0 + 16][c4] = h1;
            for (int i = tid; i < 1024; i += 128) {   // branchless clamp+select
                int r = i >> 5, c = i & 31;
                int cg = bn + c;
                int cc = cg < N ? cg : N - 1;
                float v = W[(size_t)(k0 + r) * N + cc];
                sBt[c][r] = (_Float16)(cg < N ? v : 0.f);
            }
        }
        __syncthreads();
        if (k0 + 32 < K) {
            __builtin_prefetch(&A[(size_t)(bm + (tid & 31)) * K + (k0 + 32)], 0, 3);
            __builtin_prefetch(&W[(size_t)(k0 + 32) * N + bn], 0, 3);
        }
        // ---- fragments: two b128 LDS loads each (CDNA5 16-bit layouts) ----
        const int arow = ((wave >> 1) * 16) + (lane & 15);
        const int akb  = (lane >> 4) * 8;        // lanes 16-31: K = 8..15 / 24..31
        h8 a0 = *(const h8*)&sA[arow][akb];
        h8 a1 = *(const h8*)&sA[arow][16 + akb];
        v16h af = __builtin_shufflevector(a0, a1, 0,1,2,3,4,5,6,7,8,9,10,11,12,13,14,15);
        const int bcol = ((wave & 1) * 16) + (lane & 15);
        const int bkb  = (lane >> 4) * 8;        // lanes 16-31: K = 8..15 / 24..31
        h8 b0 = *(const h8*)&sBt[bcol][bkb];
        h8 b1 = *(const h8*)&sBt[bcol][16 + bkb];
        v16h bf = __builtin_shufflevector(b0, b1, 0,1,2,3,4,5,6,7,8,9,10,11,12,13,14,15);
        acc = __builtin_amdgcn_wmma_f32_16x16x32_f16(false, af, false, bf,
                                                     (short)0, acc, false, false);
        __syncthreads();
    }
    // C/D layout: lane holds col = lane&15; VGPR v holds row v + (lane>>4)*8
    const int col   = bn + (wave & 1) * 16 + (lane & 15);
    const int rbase = bm + (wave >> 1) * 16 + (lane >> 4) * 8;
    if (nfull || col < N) {
        const float bv = bias[col];
        #pragma unroll
        for (int v = 0; v < 8; ++v) {
            float val = acc[v] + bv;
            if (relu) val = fmaxf(val, 0.f);
            C[(size_t)(rbase + v) * N + col] = val;
        }
    }
}

// ---------------------------------------------------------------------------
// Fused edge-MLP layer 3 + softmax(axis=n) + top-2.
// grid = (E, B/16); block = 128 (4 waves).  Each workgroup computes the full
// 16(batch) x 128(n) logit tile for one edge slot e (K = 256), so the softmax
// axis is local: the 64 MB edge_sparse tensor never touches HBM.
// W3 tiles stream through a DOUBLE-BUFFERED Tensor Data Mover pipeline: wave 0
// issues tile k+1 while tile k is consumed; s_wait_tensorcnt 1 retires the
// current tile (tensor ops complete in-order per wave).
// ---------------------------------------------------------------------------
__global__ void k_edge_top2(const float* __restrict__ H,   // (256,256) hidden
                            const float* __restrict__ W3,  // (256,65536)
                            const float* __restrict__ b3,  // (65536)
                            float* __restrict__ topv,      // (B*E,2)
                            int* __restrict__ snd0, int* __restrict__ rcv0)
{
    __shared__ _Float16 sA[16][40];
    __shared__ float    sBf[2][32][132];  // double-buffered f32 tile, 132-dw rows
    __shared__ float    S[16][129];
    const int e    = blockIdx.x;
    const int bm   = blockIdx.y * 16;
    const int tid  = threadIdx.x;
    const int wave = tid >> 5;
    const int lane = tid & 31;
    const int ncb  = e * N_;                   // column base in W3

#if USE_TDM
    if (tid < 32)   // prologue: DMA tile 0 into buffer 0
        tdm_load_tile(W3 + ncb, (unsigned)(unsigned long long)(void*)&sBf[0][0][0]);
#endif

    v8f acc0 = {}, acc1 = {};
    int it = 0;
    for (int k0 = 0; k0 < 256; k0 += 32, ++it) {
        // A tile 16x32: one b128 load + one b64 LDS store per thread
        {
            int r = tid >> 3, c4 = (tid & 7) << 2;
            f4 v = *(const f4*)&H[(bm + r) * 256 + k0 + c4];
            h4 h = { (_Float16)v[0], (_Float16)v[1], (_Float16)v[2], (_Float16)v[3] };
            *(h4*)&sA[r][c4] = h;
        }
#if USE_TDM
        if (tid < 32) {
            if (k0 + 32 < 256) {    // issue next tile, then retire current
                tdm_load_tile(W3 + ((size_t)(k0 + 32) * 65536 + ncb),
                              (unsigned)(unsigned long long)(void*)&sBf[(it + 1) & 1][0][0]);
                __builtin_amdgcn_s_wait_tensorcnt(1);
            } else {
                __builtin_amdgcn_s_wait_tensorcnt(0);
            }
        }
#else
        for (int s = tid; s < 1024; s += 128) {   // 32x128 f32, b128 copies
            int r = s >> 5, c4 = (s & 31) << 2;
            *(f4*)&sBf[it & 1][r][c4] = *(const f4*)&W3[(size_t)(k0 + r) * 65536 + ncb + c4];
        }
#endif
        __syncthreads();

        const float (*Bc)[132] = sBf[it & 1];
        const int arow = lane & 15;
        const int akb  = (lane >> 4) * 8;
        h8 a0 = *(const h8*)&sA[arow][akb];
        h8 a1 = *(const h8*)&sA[arow][16 + akb];
        v16h af = __builtin_shufflevector(a0, a1, 0,1,2,3,4,5,6,7,8,9,10,11,12,13,14,15);

        const int bc  = lane & 15;
        const int bkb = (lane >> 4) * 8;
        const int nb  = wave * 32;                 // this wave's 32 columns
        v16h bf0, bf1;
        #pragma unroll
        for (int i = 0; i < 16; ++i) {             // bf[j] = B[bkb+j+(j>=8?8:0)][col]
            bf0[i] = (_Float16)Bc[bkb + i + (i >= 8 ? 8 : 0)][nb + bc];
            bf1[i] = (_Float16)Bc[bkb + i + (i >= 8 ? 8 : 0)][nb + 16 + bc];
        }
        acc0 = __builtin_amdgcn_wmma_f32_16x16x32_f16(false, af, false, bf0,
                                                      (short)0, acc0, false, false);
        acc1 = __builtin_amdgcn_wmma_f32_16x16x32_f16(false, af, false, bf1,
                                                      (short)0, acc1, false, false);
        __syncthreads();
    }
    // Spill logits (+bias) to LDS
    {
        const int c0 = wave * 32 + (lane & 15);
        const int c1 = c0 + 16;
        const int rb = (lane >> 4) * 8;
        #pragma unroll
        for (int v = 0; v < 8; ++v) {
            S[rb + v][c0] = acc0[v] + b3[ncb + c0];
            S[rb + v][c1] = acc1[v] + b3[ncb + c1];
        }
    }
    __syncthreads();
    // Per-row softmax + top-2 (softmax is monotonic -> top indices from logits)
    if (tid < 16) {
        const int b = bm + tid;
        float m = -1e30f;
        for (int n = 0; n < N_; ++n) m = fmaxf(m, S[tid][n]);
        float sum = 0.f, v1 = -1e30f, v2 = -1e30f;
        int i1 = 0, i2 = 0;
        for (int n = 0; n < N_; ++n) {
            float v = S[tid][n];
            sum += __expf(v - m);
            if (v > v1)      { v2 = v1; i2 = i1; v1 = v; i1 = n; }
            else if (v > v2) { v2 = v; i2 = n; }
        }
        const long  flat = (long)b * E_ + e;
        const float inv  = 1.f / sum;
        topv[flat * 2 + 0] = __expf(v1 - m) * inv;
        topv[flat * 2 + 1] = __expf(v2 - m) * inv;
        snd0[flat] = i1 + b * N_;
        rcv0[flat] = i2 + b * N_;
    }
}

// ---------------------------------------------------------------------------
// Small helper kernels
// ---------------------------------------------------------------------------
__global__ void k_zero(float* __restrict__ p, long n) {
    long i = (long)blockIdx.x * blockDim.x + threadIdx.x;
    if (i < n) p[i] = 0.f;
}

// Zero-pad a (K,N) weight matrix to (Kpad,N) in workspace.
__global__ void k_padw(const float* __restrict__ src, float* __restrict__ dst,
                       int K, int Kpad, int N) {
    long i = (long)blockIdx.x * blockDim.x + threadIdx.x;
    if (i >= (long)Kpad * N) return;
    long r = i / N;
    int  c = (int)(i % N);
    dst[i] = (r < K) ? src[r * N + c] : 0.f;
}

__global__ void k_counts(const float* __restrict__ x, int* __restrict__ nn, int* __restrict__ ne) {
    int b = threadIdx.x;
    if (b < B_) {
        nn[b] = (int)rintf(x[b * LAT_ + (LAT_ - 2)]);
        ne[b] = (int)rintf(x[b * LAT_ + (LAT_ - 1)]);
    }
}

__global__ void k_edgef(const float* __restrict__ topv, const float* __restrict__ feats,
                        float* __restrict__ edgef) {
    long r = (long)blockIdx.x * blockDim.x + threadIdx.x;
    if (r >= BE_) return;
    edgef[r * 8 + 0] = topv[r * 2 + 0];
    edgef[r * 8 + 1] = topv[r * 2 + 1];
    #pragma unroll
    for (int f = 0; f < 6; ++f) edgef[r * 8 + 2 + f] = feats[r * 6 + f];
}

__global__ void k_gid_div(int* __restrict__ gid, long rows, int divv) {
    long r = (long)blockIdx.x * blockDim.x + threadIdx.x;
    if (r < rows) gid[r] = (int)(r / divv);
}

__global__ void k_gid_split(int* __restrict__ gid, long rows, int per,
                            const int* __restrict__ counts) {
    long r = (long)blockIdx.x * blockDim.x + threadIdx.x;
    if (r < rows) {
        int b = (int)(r / per), w = (int)(r % per);
        gid[r] = 2 * b + (w >= counts[b] ? 1 : 0);
    }
}

// e_in = [edges | nodes[snd] | nodes[rcv] | glob[gid]] with zero pad to COP cols
__global__ void k_build_ein(const float* __restrict__ edges, int CE,
                            const float* __restrict__ nodes, int CN,
                            const float* __restrict__ glob,
                            const int* __restrict__ snd, const int* __restrict__ rcv,
                            const int* __restrict__ gid,
                            float* __restrict__ out, long rows, int CO, int COP)
{
    long i = (long)blockIdx.x * blockDim.x + threadIdx.x;
    if (i >= rows * COP) return;
    long r = i / COP;
    int  c = (int)(i % COP);
    float v;
    if (c >= CO)              v = 0.f;
    else if (c < CE)          v = edges[r * CE + c];
    else if (c < CE + CN)     v = nodes[(long)snd[r] * CN + (c - CE)];
    else if (c < CE + 2 * CN) v = nodes[(long)rcv[r] * CN + (c - CE - CN)];
    else                      v = glob[(long)gid[r] * LAT_ + (c - CE - 2 * CN)];
    out[i] = v;
}

// n_in = [nodes | agg/max(cnt,1) | glob[gid]] with zero pad to COP cols
__global__ void k_build_nin(const float* __restrict__ nodes, int CN,
                            const float* __restrict__ agg, const float* __restrict__ cnt, int F,
                            const float* __restrict__ glob, const int* __restrict__ gid,
                            float* __restrict__ out, long rows, int CO, int COP)
{
    long i = (long)blockIdx.x * blockDim.x + threadIdx.x;
    if (i >= rows * COP) return;
    long r = i / COP;
    int  c = (int)(i % COP);
    float v;
    if (c >= CO)         v = 0.f;
    else if (c < CN)     v = nodes[r * CN + c];
    else if (c < CN + F) v = agg[r * F + (c - CN)] / fmaxf(cnt[r], 1.f);
    else                 v = glob[(long)gid[r] * LAT_ + (c - CN - F)];
    out[i] = v;
}

// segment sum (mean numerator) + count, with JAX-style OOB drop
__global__ void k_agg(const float* __restrict__ edges, int F, const int* __restrict__ rcv,
                      float* __restrict__ agg, float* __restrict__ cnt, long rows, int nseg)
{
    long i = (long)blockIdx.x * blockDim.x + threadIdx.x;
    if (i >= rows * F) return;
    long r = i / F;
    int  f = (int)(i % F);
    int  seg = rcv[r];
    if (seg < 0 || seg >= nseg) return;
    atomicAdd(&agg[(long)seg * F + f], edges[i]);
    if (f == 0) atomicAdd(&cnt[seg], 1.f);
}

__global__ void k_node_softmax(const float* __restrict__ pn, float* __restrict__ nprob) {
    __shared__ float red[N_];
    const int b = blockIdx.x, t = threadIdx.x;
    float v = pn[b * N_ + t];
    red[t] = v; __syncthreads();
    for (int s = 64; s > 0; s >>= 1) { if (t < s) red[t] = fmaxf(red[t], red[t + s]); __syncthreads(); }
    float m = red[0]; __syncthreads();
    float e = __expf(v - m);
    red[t] = e; __syncthreads();
    for (int s = 64; s > 0; s >>= 1) { if (t < s) red[t] += red[t + s]; __syncthreads(); }
    nprob[b * N_ + t] = e / red[0] + 1.f;
}

// Stable descending argsort of node probs (bitonic, tie -> lower index first),
// plus masked gather of sorted nodes and the inverse permutation.
__global__ void k_node_sort(const float* __restrict__ nprob, const float* __restrict__ nodef,
                            const int* __restrict__ nn, float* __restrict__ snodes,
                            int* __restrict__ invp)
{
    __shared__ float kv[N_];
    __shared__ int   ki[N_];
    const int b = blockIdx.x, t = threadIdx.x;
    kv[t] = nprob[b * N_ + t]; ki[t] = t;
    __syncthreads();
    for (int k = 2; k <= N_; k <<= 1) {
        for (int j = k >> 1; j > 0; j >>= 1) {
            int ixj = t ^ j;
            if (ixj > t) {
                bool desc = ((t & k) == 0);
                float va = kv[t], vb = kv[ixj];
                int   ia = ki[t], ib = ki[ixj];
                bool abefore = (va > vb) || (va == vb && ia < ib);
                bool sw = desc ? (!abefore) : abefore;
                if (sw) { kv[t] = vb; kv[ixj] = va; ki[t] = ib; ki[ixj] = ia; }
            }
            __syncthreads();
        }
    }
    const int src = ki[t];
    const bool m  = (t < nn[b]);
    #pragma unroll
    for (int f = 0; f < 16; ++f)
        snodes[((long)b * N_ + t) * 16 + f] = m ? nodef[((long)b * N_ + src) * 16 + f] : 0.f;
    invp[b * N_ + src] = b * N_ + t;
}

__global__ void k_map_perm(const int* __restrict__ invp, const int* __restrict__ s0,
                           const int* __restrict__ r0, int* __restrict__ ns,
                           int* __restrict__ nr, long rows) {
    long r = (long)blockIdx.x * blockDim.x + threadIdx.x;
    if (r < rows) { ns[r] = invp[s0[r]]; nr[r] = invp[r0[r]]; }
}

// Per-graph edge validity, count, stable descending sort by edge_prob,
// gather sorted edges/senders/receivers with exclusion masking.
__global__ void k_edge_sort(const int* __restrict__ nsnd, const int* __restrict__ nrcv,
                            const int* __restrict__ nn, const int* __restrict__ ne,
                            const float* __restrict__ edgef, float* __restrict__ sedges,
                            int* __restrict__ snd2, int* __restrict__ rcv2,
                            int* __restrict__ ne2out)
{
    __shared__ int kv[E_];
    __shared__ int ki[E_];
    __shared__ int cr[256];
    const int b = blockIdx.x, t = threadIdx.x;
    const int nlimit = nn[b] + b * N_;   // n_node_mod
    int c = 0;
    for (int i = t; i < E_; i += 256) {
        int r = b * E_ + i;
        bool logic = (nsnd[r] < nlimit) && (nrcv[r] < nlimit);
        kv[i] = logic ? (r + 1) : 0;
        ki[i] = i;
        if (logic) c++;
    }
    cr[t] = c;
    __syncthreads();
    for (int s = 128; s > 0; s >>= 1) { if (t < s) cr[t] += cr[t + s]; __syncthreads(); }
    const int ne2v = min(ne[b], cr[0]);
    if (t == 0) ne2out[b] = ne2v;
    for (int k = 2; k <= E_; k <<= 1) {
        for (int j = k >> 1; j > 0; j >>= 1) {
            for (int i = t; i < E_; i += 256) {
                int ixj = i ^ j;
                if (ixj > i) {
                    bool desc = ((i & k) == 0);
                    int va = kv[i], vb = kv[ixj];
                    int ia = ki[i], ib = ki[ixj];
                    bool abefore = (va > vb) || (va == vb && ia < ib);
                    bool sw = desc ? (!abefore) : abefore;
                    if (sw) { kv[i] = vb; kv[ixj] = va; ki[i] = ib; ki[ixj] = ia; }
                }
            }
            __syncthreads();
        }
    }
    for (int i = t; i < E_; i += 256) {
        int  src = ki[i];
        long rs = (long)b * E_ + src;
        long rd = (long)b * E_ + i;
        bool excl = (i >= ne2v);
        snd2[rd] = excl ? nlimit : nsnd[rs];
        rcv2[rd] = excl ? nlimit : nrcv[rs];
        #pragma unroll
        for (int f = 0; f < 8; ++f)
            sedges[rd * 8 + f] = excl ? 0.f : edgef[rs * 8 + f];
    }
}

__global__ void k_newglob(const float* __restrict__ x, float* __restrict__ g) {
    long i = (long)blockIdx.x * blockDim.x + threadIdx.x;
    if (i < 2L * B_ * LAT_) {
        long r = i >> 6;
        int  c = (int)(i & 63);
        g[i] = (r & 1) ? 0.f : x[(r >> 1) * LAT_ + c];
    }
}

__global__ void k_i2f(float* __restrict__ dst, const int* __restrict__ src, long n) {
    long i = (long)blockIdx.x * blockDim.x + threadIdx.x;
    if (i < n) dst[i] = (float)src[i];
}

__global__ void k_write_counts(float* __restrict__ nnode, float* __restrict__ nedge,
                               const int* __restrict__ nn, const int* __restrict__ ne2) {
    int b = threadIdx.x;
    if (b < B_) {
        nnode[2 * b]     = (float)nn[b];
        nnode[2 * b + 1] = (float)(N_ - nn[b]);
        nedge[2 * b]     = (float)ne2[b];
        nedge[2 * b + 1] = (float)(E_ - ne2[b]);
    }
}

// ---------------------------------------------------------------------------
extern "C" void kernel_launch(void* const* d_in, const int* in_sizes, int n_in,
                              void* d_out, int out_size, void* d_ws, size_t ws_size,
                              hipStream_t stream)
{
    (void)in_sizes; (void)n_in; (void)out_size; (void)ws_size;
    const float* x    = (const float*)d_in[0];
    const float* We1  = (const float*)d_in[1];  const float* be1  = (const float*)d_in[2];
    const float* We2  = (const float*)d_in[3];  const float* be2  = (const float*)d_in[4];
    const float* We3  = (const float*)d_in[5];  const float* be3  = (const float*)d_in[6];
    const float* Wf1  = (const float*)d_in[7];  const float* bf1  = (const float*)d_in[8];
    const float* Wf2  = (const float*)d_in[9];  const float* bf2  = (const float*)d_in[10];
    const float* Wf3  = (const float*)d_in[11]; const float* bf3  = (const float*)d_in[12];
    const float* Wn1  = (const float*)d_in[13]; const float* bn1  = (const float*)d_in[14];
    const float* Wn2  = (const float*)d_in[15]; const float* bn2  = (const float*)d_in[16];
    const float* Wn3  = (const float*)d_in[17]; const float* bn3  = (const float*)d_in[18];
    const float* Wpe1 = (const float*)d_in[19]; const float* bpe1 = (const float*)d_in[20];
    const float* Wpn1 = (const float*)d_in[21]; const float* bpn1 = (const float*)d_in[22];
    const float* Wpe2 = (const float*)d_in[23]; const float* bpe2 = (const float*)d_in[24];
    const float* Wpn2 = (const float*)d_in[25]; const float* bpn2 = (const float*)d_in[26];
    const float* Wfe1 = (const float*)d_in[27]; const float* bfe1 = (const float*)d_in[28];
    const float* Wfe2 = (const float*)d_in[29]; const float* bfe2 = (const float*)d_in[30];
    const float* Wfn1 = (const float*)d_in[31]; const float* bfn1 = (const float*)d_in[32];
    const float* Wfn2 = (const float*)d_in[33]; const float* bfn2 = (const float*)d_in[34];

    // Output layout (tuple concatenated flat, all as f32)
    float* out     = (float*)d_out;
    float* fn_o    = out;               // (B*N,16)  524288
    float* fe_o    = out + 524288;      // (B*E,8)  1048576
    float* snd_o   = out + 1572864;     // (B*E)
    float* rcv_o   = out + 1703936;     // (B*E)
    float* nnode_o = out + 1835008;     // (2B)
    float* nedge_o = out + 1835520;     // (2B)
    float* glob_o  = out + 1836032;     // (2B,64)

    // Workspace carve (bump allocator; large concat buffers are reused)
    float* wsf = (float*)d_ws;
    size_t off = 0;
    auto alloc = [&](size_t n) { float* p = wsf + off; off += n; return p; };

    float* h1     = alloc(65536);
    float* h2     = alloc(65536);
    float* topv   = alloc((size_t)BE_ * 2);
    int*   snd0   = (int*)alloc(BE_);
    int*   rcv0   = (int*)alloc(BE_);
    float* feats  = alloc((size_t)BE_ * 6);
    float* nodef  = alloc((size_t)BN_ * 16);
    float* edgef  = alloc((size_t)BE_ * 8);
    int*   nn     = (int*)alloc(B_);
    int*   ne     = (int*)alloc(B_);
    int*   ne2    = (int*)alloc(B_);
    int*   egid1  = (int*)alloc(BE_);
    int*   ngid1  = (int*)alloc(BN_);
    float* big    = alloc((size_t)BE_ * 256);  // e_in (128 / 256 / 128 padded) reuse
    float* ehid   = alloc((size_t)BE_ * 64);   // edges1, later final edge hidden
    float* agg    = alloc((size_t)BN_ * 64);
    float* cnt    = alloc(BN_);
    float* nin    = alloc((size_t)BN_ * 160);  // n_in (160 / 160 / 96 padded) reuse
    float* nhid   = alloc((size_t)BN_ * 64);   // nodes1, later final node hidden
    float* edges2 = alloc(BE_);
    float* agg2   = alloc(BN_);
    float* cnt2   = alloc(BN_);
    float* pnbuf  = alloc(BN_);
    float* nprob  = alloc(BN_);
    int*   invp   = (int*)alloc(BN_);
    float* snodes = alloc((size_t)BN_ * 16);
    int*   nsnd   = (int*)alloc(BE_);
    int*   nrcv   = (int*)alloc(BE_);
    float* sedges = alloc((size_t)BE_ * 8);
    int*   snd2   = (int*)alloc(BE_);
    int*   rcv2   = (int*)alloc(BE_);
    int*   egid2  = (int*)alloc(BE_);
    int*   ngid2  = (int*)alloc(BN_);
    float* fagg   = alloc((size_t)BN_ * 8);
    float* fcnt   = alloc(BN_);
    // K-padded weight copies (zero pad rows)
    float* Wpe1p  = alloc(128 * 64);
    float* Wpn1p  = alloc(160 * 64);
    float* Wpn2p  = alloc(160 * 1);
    float* Wfe1p  = alloc(128 * 64);
    float* Wfn1p  = alloc(96 * 64);

    auto gemm = [&](const float* A, const float* W, const float* b, float* C,
                    int M, int Ncol, int Kpad, int relu) {
        dim3 grid((Ncol + 31) / 32, (M + 31) / 32);
        k_gemm<<<grid, 128, 0, stream>>>(A, W, b, C, M, Ncol, Kpad, relu);
    };
    auto ew = [](long n) { return dim3((unsigned)((n + 255) / 256)); };

    k_counts<<<1, 256, 0, stream>>>(x, nn, ne);
    k_padw<<<ew(128 * 64), 256, 0, stream>>>(Wpe1, Wpe1p, 104, 128, 64);
    k_padw<<<ew(160 * 64), 256, 0, stream>>>(Wpn1, Wpn1p, 144, 160, 64);
    k_padw<<<ew(160),      256, 0, stream>>>(Wpn2, Wpn2p, 129, 160, 1);
    k_padw<<<ew(128 * 64), 256, 0, stream>>>(Wfe1, Wfe1p, 104, 128, 64);
    k_padw<<<ew(96 * 64),  256, 0, stream>>>(Wfn1, Wfn1p, 88, 96, 64);

    // Front MLPs (first two layers = 256x256 GEMMs w/ ReLU)
    gemm(x,  We1, be1, h1, B_, 256, LAT_, 1);
    gemm(h1, We2, be2, h2, B_, 256, 256, 1);
    k_edge_top2<<<dim3(E_, B_ / 16), 128, 0, stream>>>(h2, We3, be3, topv, snd0, rcv0);

    gemm(x,  Wf1, bf1, h1, B_, 256, LAT_, 1);
    gemm(h1, Wf2, bf2, h2, B_, 256, 256, 1);
    gemm(h2, Wf3, bf3, feats, B_, E_ * 6, 256, 0);

    gemm(x,  Wn1, bn1, h1, B_, 256, LAT_, 1);
    gemm(h1, Wn2, bn2, h2, B_, 256, 256, 1);
    gemm(h2, Wn3, bn3, nodef, B_, N_ * 16, 256, 0);

    k_edgef<<<ew(BE_), 256, 0, stream>>>(topv, feats, edgef);
    k_gid_div<<<ew(BE_), 256, 0, stream>>>(egid1, BE_, E_);
    k_gid_div<<<ew(BN_), 256, 0, stream>>>(ngid1, BN_, N_);

    // ---- prob GNN, layer 1 ----
    k_build_ein<<<ew((long)BE_ * 128), 256, 0, stream>>>(edgef, 8, nodef, 16, x,
                                                         snd0, rcv0, egid1, big, BE_, 104, 128);
    gemm(big, Wpe1p, bpe1, ehid, BE_, 64, 128, 0);
    k_zero<<<ew((long)BN_ * 64), 256, 0, stream>>>(agg, (long)BN_ * 64);
    k_zero<<<ew(BN_), 256, 0, stream>>>(cnt, BN_);
    k_agg<<<ew((long)BE_ * 64), 256, 0, stream>>>(ehid, 64, rcv0, agg, cnt, BE_, BN_);
    k_build_nin<<<ew((long)BN_ * 160), 256, 0, stream>>>(nodef, 16, agg, cnt, 64, x,
                                                         ngid1, nin, BN_, 144, 160);
    gemm(nin, Wpn1p, bpn1, nhid, BN_, 64, 160, 0);

    // ---- prob GNN, layer 2 ----
    k_build_ein<<<ew((long)BE_ * 256), 256, 0, stream>>>(ehid, 64, nhid, 64, x,
                                                         snd0, rcv0, egid1, big, BE_, 256, 256);
    gemm(big, Wpe2, bpe2, edges2, BE_, 1, 256, 0);
    k_zero<<<ew(BN_), 256, 0, stream>>>(agg2, BN_);
    k_zero<<<ew(BN_), 256, 0, stream>>>(cnt2, BN_);
    k_agg<<<ew(BE_), 256, 0, stream>>>(edges2, 1, rcv0, agg2, cnt2, BE_, BN_);
    k_build_nin<<<ew((long)BN_ * 160), 256, 0, stream>>>(nhid, 64, agg2, cnt2, 1, x,
                                                         ngid1, nin, BN_, 129, 160);
    gemm(nin, Wpn2p, bpn2, pnbuf, BN_, 1, 160, 0);

    // ---- sorting / permutation section ----
    k_node_softmax<<<B_, 128, 0, stream>>>(pnbuf, nprob);
    k_node_sort<<<B_, 128, 0, stream>>>(nprob, nodef, nn, snodes, invp);
    k_map_perm<<<ew(BE_), 256, 0, stream>>>(invp, snd0, rcv0, nsnd, nrcv, BE_);
    k_edge_sort<<<B_, 256, 0, stream>>>(nsnd, nrcv, nn, ne, edgef, sedges, snd2, rcv2, ne2);
    k_newglob<<<ew(2L * B_ * LAT_), 256, 0, stream>>>(x, glob_o);

    // ---- final GNN (1 layer, 2-layer MLPs with ReLU) ----
    k_gid_split<<<ew(BE_), 256, 0, stream>>>(egid2, BE_, E_, ne2);
    k_gid_split<<<ew(BN_), 256, 0, stream>>>(ngid2, BN_, N_, nn);
    k_build_ein<<<ew((long)BE_ * 128), 256, 0, stream>>>(sedges, 8, snodes, 16, glob_o,
                                                         snd2, rcv2, egid2, big, BE_, 104, 128);
    gemm(big,  Wfe1p, bfe1, ehid, BE_, 64, 128, 1);
    gemm(ehid, Wfe2, bfe2, fe_o, BE_, 8, 64, 0);
    k_zero<<<ew((long)BN_ * 8), 256, 0, stream>>>(fagg, (long)BN_ * 8);
    k_zero<<<ew(BN_), 256, 0, stream>>>(fcnt, BN_);
    k_agg<<<ew((long)BE_ * 8), 256, 0, stream>>>(fe_o, 8, rcv2, fagg, fcnt, BE_, BN_);
    k_build_nin<<<ew((long)BN_ * 96), 256, 0, stream>>>(snodes, 16, fagg, fcnt, 8, glob_o,
                                                        ngid2, nin, BN_, 88, 96);
    gemm(nin,  Wfn1p, bfn1, nhid, BN_, 64, 96, 1);
    gemm(nhid, Wfn2, bfn2, fn_o, BN_, 16, 64, 0);

    // ---- tail outputs ----
    k_i2f<<<ew(BE_), 256, 0, stream>>>(snd_o, snd2, BE_);
    k_i2f<<<ew(BE_), 256, 0, stream>>>(rcv_o, rcv2, BE_);
    k_write_counts<<<1, 256, 0, stream>>>(nnode_o, nedge_o, nn, ne2);
}
